// HeteroGNNWithAttention_32581621907695
// MI455X (gfx1250) — matrix operand order repaired and checked by reading, not compiled, not run on hardware
//
#include <hip/hip_runtime.h>
#include <hip/hip_bf16.h>

// ---------------------------------------------------------------------------
// Hetero-GATv2 on MI455X (gfx1250). Dense 64->256 projections run on the
// bf16 WMMA pipe (v_wmma_f32_16x16x32_bf16, fp32 accumulate) with the A tile
// staged through LDS (ds_load_b128 fragments) and weights pre-converted to a
// transposed bf16 buffer (global_load_b128 fragments). Edge softmax/scatter
// phases are memory-bound float-atomic work. All launches on `stream`;
// scratch in d_ws; zero-fills via hipMemsetAsync (graph-capturable).
// ---------------------------------------------------------------------------

#define N_IP    50000
#define N_PORT  10000
#define N_PROTO 256
#define C       64
#define H       4
#define HC      256   // H*C
#define NEG_SLOPE 0.2f

typedef __attribute__((ext_vector_type(16))) __bf16 v16bf;
typedef __attribute__((ext_vector_type(8)))  __bf16 v8bf;
typedef __attribute__((ext_vector_type(8)))  float  v8f;

// ================== weight convert + transpose (once per GEMM) =============
// Wt[n*64 + k] = bf16(W[k*256 + n])  -- makes each B fragment two b128 loads
__global__ __launch_bounds__(256)
void cvt_w_bf16(const float* __restrict__ W, __bf16* __restrict__ Wt) {
  int t = blockIdx.x * 256 + threadIdx.x;      // 16384 elements
  int k = t >> 8, n = t & 255;
  Wt[n * C + k] = (__bf16)W[t];
}

// ============================ WMMA GEMM ====================================
// Y[M,256] = bf16(X[M,64]) @ bf16(W[64,256]) + bias[256]
// Block = 256 threads = 8 waves; covers 16 rows x 128 cols; grid = (M/16, 2).
// A tile (16x64) is loaded once per block, converted to bf16 in LDS, then each
// wave pulls its WMMA fragments with ds_load_b128. K=64 -> two WMMA steps.
__global__ __launch_bounds__(256)
void gemm_wmma_64x256(const float* __restrict__ X, const __bf16* __restrict__ Wt,
                      const float* __restrict__ bias, float* __restrict__ Y) {
  __shared__ __bf16 sA[16 * C];                // 2 KB, row-major [m][k]

  const int bm = blockIdx.x << 4;

  // ---- stage A tile: 1024 floats, 4 per thread, coalesced b128 reads ----
  {
    const int idx = threadIdx.x << 2;          // 0,4,...,1020
    const int m = idx >> 6, cc = idx & 63;
    const float4 v = *(const float4*)(X + (size_t)(bm + m) * C + cc);
    union { __bf16 b[4]; uint2 u; } pk;
    pk.b[0] = (__bf16)v.x; pk.b[1] = (__bf16)v.y;
    pk.b[2] = (__bf16)v.z; pk.b[3] = (__bf16)v.w;
    ((uint2*)sA)[threadIdx.x] = pk.u;          // ds_store_b64, 8B aligned
  }
  __syncthreads();

  const int wave = threadIdx.x >> 5;
  const int lane = threadIdx.x & 31;
  const int lm   = lane & 15;                  // row (A) / col (B,C,D) in tile
  const int kh   = lane >> 4;                  // lane-half -> K sub-block
  const int bn   = (blockIdx.y << 7) + (wave << 4);
  const int n    = bn + lm;
  __builtin_prefetch(Wt + (size_t)n * C, 0, 1);    // global_prefetch_b8

  v8f acc = {};
#pragma unroll
  for (int ks = 0; ks < 2; ++ks) {
    // A k order used here (contiguous, k = 16*kh + s): lanes0-15: elems0-15=K0-15;
    //                      lanes16-31: elems0-15=K16-31
    const int k0 = (ks << 5) + (kh << 4);
    const v8bf alo = *(const v8bf*)(sA + lm * C + k0);        // ds_load_b128
    const v8bf ahi = *(const v8bf*)(sA + lm * C + k0 + 8);    // ds_load_b128
    v16bf a;
#pragma unroll
    for (int i = 0; i < 8; ++i) { a[i] = alo[i]; a[8 + i] = ahi[i]; }

    // B uses the same contiguous k order as A: lanes0-15 hold K=0..15, lanes16-31 hold K=16..31
    const __bf16* wp = Wt + (size_t)n * C + (ks << 5) + (kh << 4);
    const v8bf blo = *(const v8bf*)(wp);                      // global_load_b128
    const v8bf bhi = *(const v8bf*)(wp + 8);                  // global_load_b128
    v16bf b;
#pragma unroll
    for (int i = 0; i < 8; ++i) { b[i] = blo[i]; b[8 + i] = bhi[i]; }

    acc = __builtin_amdgcn_wmma_f32_16x16x32_bf16(false, a, false, b,
                                                  (short)0, acc, false, false);
  }

  const float bv = bias[n];
  // C/D layout: VGPR r -> M = r + 8*kh, N = lane%16
#pragma unroll
  for (int r = 0; r < 8; ++r) {
    const int m = bm + r + (kh << 3);
    Y[(size_t)m * HC + n] = acc[r] + bv;
  }
}

// ========================= input embeddings ================================
__global__ __launch_bounds__(256)
void embed_kernel(const float* __restrict__ x, const float* __restrict__ W,
                  const float* __restrict__ b, float* __restrict__ h,
                  int N, int K) {
  int t = blockIdx.x * blockDim.x + threadIdx.x;
  if (t >= N * C) return;
  int n = t >> 6, c = t & 63;
  float acc = b[c];
  const float* xr = x + (size_t)n * K;
  for (int k = 0; k < K; ++k) acc += xr[k] * W[k * C + c];
  h[t] = acc;
}

// ========================= edge phase ======================================
// logits[e,h] = att[h] . leakyrelu(xl[src]+xr[dst]); atomic segment-max via
// monotone unsigned key (neg -> ~u, pos -> u|msb).
__global__ __launch_bounds__(256)
void edge_logits(const float* __restrict__ xl, const float* __restrict__ xr,
                 const int* __restrict__ src, const int* __restrict__ dst,
                 const float* __restrict__ att, float* __restrict__ logit,
                 unsigned int* __restrict__ mkey, int E) {
  int t = blockIdx.x * blockDim.x + threadIdx.x;
  if (t >= E * H) return;
  int e = t >> 2, h = t & 3;
  int s = src[e], d = dst[e];
  const float4* pl = (const float4*)(xl + (size_t)s * HC + h * C);
  const float4* pr = (const float4*)(xr + (size_t)d * HC + h * C);
  const float4* pa = (const float4*)(att + h * C);
  float acc = 0.f;
#pragma unroll 4
  for (int c4 = 0; c4 < C / 4; ++c4) {
    float4 a = pl[c4], r = pr[c4], w = pa[c4];
    float v;
    v = a.x + r.x; v = v > 0.f ? v : NEG_SLOPE * v; acc += w.x * v;
    v = a.y + r.y; v = v > 0.f ? v : NEG_SLOPE * v; acc += w.y * v;
    v = a.z + r.z; v = v > 0.f ? v : NEG_SLOPE * v; acc += w.z * v;
    v = a.w + r.w; v = v > 0.f ? v : NEG_SLOPE * v; acc += w.w * v;
  }
  logit[t] = acc;
  unsigned int u = __float_as_uint(acc);
  u = (u & 0x80000000u) ? ~u : (u | 0x80000000u);
  atomicMax(mkey + (size_t)d * H + h, u);
}

// ex = exp(logit - segmax); denom[dst,h] += ex  (ex overwrites logit buffer)
__global__ __launch_bounds__(256)
void edge_exp(float* __restrict__ logit_ex, const unsigned int* __restrict__ mkey,
              float* __restrict__ denom, const int* __restrict__ dst, int E) {
  int t = blockIdx.x * blockDim.x + threadIdx.x;
  if (t >= E * H) return;
  int e = t >> 2, h = t & 3;
  int d = dst[e];
  unsigned int u = mkey[(size_t)d * H + h];
  unsigned int bits = (u & 0x80000000u) ? (u & 0x7FFFFFFFu) : ~u;
  float m = __uint_as_float(bits);
  float ex = expf(logit_ex[t] - m);
  logit_ex[t] = ex;
  atomicAdd(denom + (size_t)d * H + h, ex);
}

// acc[dst,h,c] += (ex/denom) * xl[src,h,c]; one block (256 thr) per edge
__global__ __launch_bounds__(256)
void edge_scatter(const float* __restrict__ xl, const float* __restrict__ ex,
                  const float* __restrict__ denom, const int* __restrict__ src,
                  const int* __restrict__ dst, float* __restrict__ acc, int E) {
  int e = blockIdx.x;
  if (e >= E) return;
  int idx = threadIdx.x;          // 0..255 -> (h = idx/64, c = idx%64)
  int h = idx >> 6;
  int s = src[e], d = dst[e];
  float alpha = ex[(size_t)e * H + h] / (denom[(size_t)d * H + h] + 1e-16f);
  atomicAdd(acc + (size_t)d * HC + idx, alpha * xl[(size_t)s * HC + idx]);
}

// h_out = elu( scale * ( mean_heads(acc) + b1 (+ b2) ) )
__global__ __launch_bounds__(256)
void combine_kernel(const float* __restrict__ acc, const float* __restrict__ b1,
                    const float* __restrict__ b2, float scale,
                    float* __restrict__ hout, int N) {
  int t = blockIdx.x * blockDim.x + threadIdx.x;
  if (t >= N * C) return;
  int n = t >> 6, c = t & 63;
  const float* p = acc + (size_t)n * HC + c;
  float s = 0.25f * (p[0] + p[64] + p[128] + p[192]);
  s += b1[c];
  if (b2) s += b2[c];
  s *= scale;
  hout[t] = s > 0.f ? s : (expf(s) - 1.f);     // elu
}

// ========================= pooling / classifier ============================
__global__ __launch_bounds__(256)
void pool_scores(const float* __restrict__ h, const float* __restrict__ W1,
                 const float* __restrict__ b1, const float* __restrict__ W2,
                 const float* __restrict__ b2, float* __restrict__ scores, int N) {
  int n = blockIdx.x * blockDim.x + threadIdx.x;
  if (n >= N) return;
  const float* hp = h + (size_t)n * C;
  float s = b2[0];
  for (int j = 0; j < 32; ++j) {
    float t = b1[j];
    for (int c = 0; c < C; ++c) t += hp[c] * W1[c * 32 + j];
    s += tanhf(t) * W2[j];
  }
  scores[n] = s;
}

__global__ __launch_bounds__(256)
void reduce_max(const float* __restrict__ scores, int N, float* __restrict__ red) {
  __shared__ float sm[256];
  float m = -3.4e38f;
  for (int i = threadIdx.x; i < N; i += 256) m = fmaxf(m, scores[i]);
  sm[threadIdx.x] = m;
  __syncthreads();
  for (int s = 128; s > 0; s >>= 1) {
    if (threadIdx.x < s) sm[threadIdx.x] = fmaxf(sm[threadIdx.x], sm[threadIdx.x + s]);
    __syncthreads();
  }
  if (threadIdx.x == 0) red[0] = sm[0];
}

__global__ __launch_bounds__(256)
void reduce_sumexp(const float* __restrict__ scores, int N, float* __restrict__ red) {
  __shared__ float sm[256];
  float mx = red[0];
  float a = 0.f;
  for (int i = threadIdx.x; i < N; i += 256) a += expf(scores[i] - mx);
  sm[threadIdx.x] = a;
  __syncthreads();
  for (int s = 128; s > 0; s >>= 1) {
    if (threadIdx.x < s) sm[threadIdx.x] += sm[threadIdx.x + s];
    __syncthreads();
  }
  if (threadIdx.x == 0) red[1] = sm[0];
}

// emb[c] = sum_n softmax(score)_n * h[n,c]; one block per channel
__global__ __launch_bounds__(256)
void pool_weighted(const float* __restrict__ h, const float* __restrict__ scores,
                   const float* __restrict__ red, float* __restrict__ emb, int N) {
  __shared__ float sm[256];
  int c = blockIdx.x;
  float mx = red[0], sum = red[1];
  float a = 0.f;
  for (int i = threadIdx.x; i < N; i += 256)
    a += (expf(scores[i] - mx) / sum) * h[(size_t)i * C + c];
  sm[threadIdx.x] = a;
  __syncthreads();
  for (int s = 128; s > 0; s >>= 1) {
    if (threadIdx.x < s) sm[threadIdx.x] += sm[threadIdx.x + s];
    __syncthreads();
  }
  if (threadIdx.x == 0) emb[c] = sm[0];
}

__global__ void final_clf(const float* __restrict__ emb, const float* __restrict__ W1,
                          const float* __restrict__ b1, const float* __restrict__ W2,
                          const float* __restrict__ b2, float* __restrict__ out) {
  __shared__ float hid[32];
  int j = threadIdx.x;
  if (j < 32) {
    float t = b1[j];
    for (int c = 0; c < C; ++c) t += emb[c] * W1[c * 32 + j];
    hid[j] = t > 0.f ? t : 0.f;  // relu
  }
  __syncthreads();
  if (j < 2) {
    float o = b2[j];
    for (int k = 0; k < 32; ++k) o += hid[k] * W2[k * 2 + j];
    out[j] = o;
  }
}

// ============================ driver =======================================
extern "C" void kernel_launch(void* const* d_in, const int* in_sizes, int n_in,
                              void* d_out, int out_size, void* d_ws, size_t ws_size,
                              hipStream_t stream) {
  (void)in_sizes; (void)n_in; (void)out_size; (void)ws_size;
  // ---- raw inputs (setup_inputs dict order) ----
  const float* x_ip    = (const float*)d_in[0];
  const float* x_port  = (const float*)d_in[1];
  const float* x_proto = (const float*)d_in[2];
  // ---- params: jax pytree (sorted-key) flatten order ----
  // clf, ip_lin, layers[0..1]{conn,runs,served,usedby,uses}{Wl,Wr,att,b,bl,br},
  // pool, port_lin, proto_lin
  const int P_CLF = 13, P_IP = 17, P_L0 = 19, P_POOL = 79, P_PORT = 83, P_PROTO = 85;
  const float* W1c = (const float*)d_in[P_CLF + 0];
  const float* b1c = (const float*)d_in[P_CLF + 1];
  const float* W2c = (const float*)d_in[P_CLF + 2];
  const float* b2c = (const float*)d_in[P_CLF + 3];
  const float* Wip = (const float*)d_in[P_IP + 0];
  const float* bip = (const float*)d_in[P_IP + 1];
  const float* W1p = (const float*)d_in[P_POOL + 0];
  const float* b1p = (const float*)d_in[P_POOL + 1];
  const float* W2p = (const float*)d_in[P_POOL + 2];
  const float* b2p = (const float*)d_in[P_POOL + 3];
  const float* Wpo = (const float*)d_in[P_PORT + 0];
  const float* bpo = (const float*)d_in[P_PORT + 1];
  const float* Wpr = (const float*)d_in[P_PROTO + 0];
  const float* bpr = (const float*)d_in[P_PROTO + 1];

  // ---- workspace layout ----
  const size_t H_TOT   = (size_t)(N_IP + N_PORT + N_PROTO) * C;
  const size_t ACC_TOT = (size_t)(N_IP + N_PORT + N_PROTO) * HC;
  char* base = (char*)d_ws;
  size_t off = 0;
  auto take = [&](size_t nbytes) {
    char* p = base + off;
    off += (nbytes + 255) & ~(size_t)255;
    return p;
  };
  float* hc  = (float*)take(H_TOT * 4);
  float* hn  = (float*)take(H_TOT * 4);
  float* xl  = (float*)take((size_t)N_IP * HC * 4);
  float* xr  = (float*)take((size_t)N_IP * HC * 4);
  float* acc = (float*)take(ACC_TOT * 4);
  float* exb = (float*)take((size_t)400000 * H * 4);
  unsigned int* mkey = (unsigned int*)take((size_t)N_IP * H * 4);
  float* denom = (float*)take((size_t)N_IP * H * 4);
  float* scores = (float*)take((size_t)N_IP * 4);
  float* red = (float*)take(2 * 4);
  float* emb = (float*)take(C * 4);
  __bf16* wt = (__bf16*)take((size_t)C * HC * 2);   // transposed bf16 weights

  const int   n_of[3]  = {N_IP, N_PORT, N_PROTO};
  const size_t h_off[3]  = {0, (size_t)N_IP * C,  (size_t)(N_IP + N_PORT) * C};
  const size_t a_off[3]  = {0, (size_t)N_IP * HC, (size_t)(N_IP + N_PORT) * HC};

  // ---- input embeddings ----
  embed_kernel<<<(N_IP * C + 255) / 256, 256, 0, stream>>>(x_ip, Wip, bip, hc + h_off[0], N_IP, 12);
  embed_kernel<<<(N_PORT * C + 255) / 256, 256, 0, stream>>>(x_port, Wpo, bpo, hc + h_off[1], N_PORT, 6);
  embed_kernel<<<(N_PROTO * C + 255) / 256, 256, 0, stream>>>(x_proto, Wpr, bpr, hc + h_off[2], N_PROTO, 5);

  // edge types: {src_type, dst_type, E, src_idx, dst_idx, param_off}
  struct ET { int st, dt, E, si, di, poff; };
  const ET ets[5] = {
    {0, 0, 400000,  3,  4,  0},   // conn   (ip  -> ip)
    {0, 1, 400000,  5,  6, 24},   // uses   (ip  -> port)
    {1, 2, 100000,  7,  8,  6},   // runs   (port-> proto)
    {1, 0, 400000,  9, 10, 18},   // usedby (port-> ip)
    {2, 1, 100000, 11, 12, 12},   // served (proto->port)
  };

  for (int li = 0; li < 2; ++li) {
    const int base_p = P_L0 + 30 * li;
    hipMemsetAsync(acc, 0, ACC_TOT * 4, stream);
    for (int ei = 0; ei < 5; ++ei) {
      const ET& et = ets[ei];
      const int pb = base_p + et.poff;
      const float* Wl  = (const float*)d_in[pb + 0];
      const float* Wr  = (const float*)d_in[pb + 1];
      const float* att = (const float*)d_in[pb + 2];
      const float* bl  = (const float*)d_in[pb + 4];
      const float* br  = (const float*)d_in[pb + 5];
      const int*   src = (const int*)d_in[et.si];
      const int*   dst = (const int*)d_in[et.di];
      const int Ns = n_of[et.st], Nd = n_of[et.dt], E = et.E;
      // dense projections on the WMMA pipe (stream order serializes cvt/gemm)
      cvt_w_bf16<<<64, 256, 0, stream>>>(Wl, wt);
      gemm_wmma_64x256<<<dim3(Ns / 16, 2), 256, 0, stream>>>(hc + h_off[et.st], wt, bl, xl);
      cvt_w_bf16<<<64, 256, 0, stream>>>(Wr, wt);
      gemm_wmma_64x256<<<dim3(Nd / 16, 2), 256, 0, stream>>>(hc + h_off[et.dt], wt, br, xr);
      // per-dst softmax over incoming edges
      hipMemsetAsync(mkey,  0, (size_t)Nd * H * 4, stream);
      hipMemsetAsync(denom, 0, (size_t)Nd * H * 4, stream);
      edge_logits<<<(E * H + 255) / 256, 256, 0, stream>>>(xl, xr, src, dst, att, exb, mkey, E);
      edge_exp<<<(E * H + 255) / 256, 256, 0, stream>>>(exb, mkey, denom, dst, E);
      edge_scatter<<<E, 256, 0, stream>>>(xl, exb, denom, src, dst, acc + a_off[et.dt], E);
    }
    // HeteroConv mean over edge types sharing a dst + head-mean + bias + elu
    const float* b_conn   = (const float*)d_in[base_p +  0 + 3];
    const float* b_runs   = (const float*)d_in[base_p +  6 + 3];
    const float* b_served = (const float*)d_in[base_p + 12 + 3];
    const float* b_usedby = (const float*)d_in[base_p + 18 + 3];
    const float* b_uses   = (const float*)d_in[base_p + 24 + 3];
    combine_kernel<<<(N_IP * C + 255) / 256, 256, 0, stream>>>(acc + a_off[0], b_conn, b_usedby, 0.5f, hn + h_off[0], N_IP);
    combine_kernel<<<(N_PORT * C + 255) / 256, 256, 0, stream>>>(acc + a_off[1], b_uses, b_served, 0.5f, hn + h_off[1], N_PORT);
    combine_kernel<<<(N_PROTO * C + 255) / 256, 256, 0, stream>>>(acc + a_off[2], b_runs, (const float*)nullptr, 1.0f, hn + h_off[2], N_PROTO);
    float* t = hc; hc = hn; hn = t;
  }

  // ---- attention pooling over ip nodes + classifier ----
  pool_scores<<<(N_IP + 255) / 256, 256, 0, stream>>>(hc + h_off[0], W1p, b1p, W2p, b2p, scores, N_IP);
  reduce_max<<<1, 256, 0, stream>>>(scores, N_IP, red);
  reduce_sumexp<<<1, 256, 0, stream>>>(scores, N_IP, red);
  pool_weighted<<<C, 256, 0, stream>>>(hc + h_off[0], scores, red, emb, N_IP);
  final_clf<<<1, 64, 0, stream>>>(emb, W1c, b1c, W2c, b2c, (float*)d_out);
}
